// gat_hetero_22900765623056
// MI455X (gfx1250) — compile-verified, matrix operand order checked by reading, not run on hardware
//
#include <hip/hip_runtime.h>
#include <hip/hip_bf16.h>
#include <math.h>

typedef __bf16 bf16_t;
typedef __attribute__((ext_vector_type(16))) __bf16 v16bf;
typedef __attribute__((ext_vector_type(8)))  __bf16 v8bf;
typedef __attribute__((ext_vector_type(4)))  __bf16 v4bf;
typedef __attribute__((ext_vector_type(8)))  float  v8f;
typedef __attribute__((ext_vector_type(4)))  unsigned int v4u;
typedef __attribute__((ext_vector_type(8)))  int v8i;
typedef __attribute__((ext_vector_type(4)))  int v4i;

#define D 256
#define TT 4
#define NGB 64
#define NH 8
#define PS_ALPHA 0.2f
#define LN_EPS 1e-5f

// ---------------------------------------------------------------------------
// Kernel 0: one-time weight prep: fp32 [T,D,D] row-major -> bf16 transposed
// [T][n*D+k] so WMMA B fragments are contiguous 32B runs.
// ---------------------------------------------------------------------------
__global__ void __launch_bounds__(256)
transpose_cvt_weights(const float* __restrict__ W, bf16_t* __restrict__ WT, int total)
{
    int e = blockIdx.x * 256 + threadIdx.x;
    if (e < total) {
        int t = e >> 16;             // D*D == 65536
        int r = e & 65535;
        int k = r >> 8, n = r & 255;
        WT[(size_t)t * D * D + n * D + k] = (bf16_t)W[e];
    }
}

// ---------------------------------------------------------------------------
// Kernel 1: per-type compacted row projection  Y[r,:] = X[r,:] @ W[type[r]]
// WT: bf16 pre-transposed weights [T][n*D+k]. W[t]^T is DMA'd into LDS by the
// Tensor Data Mover (128KB tile) while the block builds the per-type row list;
// s_wait_tensorcnt + barrier before the WMMA tile loop consumes it.
// ---------------------------------------------------------------------------
__global__ void __launch_bounds__(256)
project_rows(const float* __restrict__ X, const int* __restrict__ types,
             const bf16_t* __restrict__ WT, bf16_t* __restrict__ Y, int rows)
{
    extern __shared__ char smem[];
    bf16_t* wT  = (bf16_t*)smem;                     // [D*D] W[t]^T bf16 (LDS offset 0)
    bf16_t* aT  = (bf16_t*)(smem + D * D * 2);       // [16*D] A tile bf16
    int* list   = (int*)(smem + D * D * 2 + 16 * D * 2); // [256]
    int* cntp   = list + 256;

    const int tid  = threadIdx.x;
    const int lane = tid & 31;
    const int wave = tid >> 5;
    const int base = blockIdx.x * 256;

    for (int t = 0; t < TT; ++t) {
        // ---- wave 0 issues TDM: 128KB of W[t]^T (bf16) global -> LDS ----
        if (tid < 32) {
            unsigned long long ga =
                (unsigned long long)(uintptr_t)(WT + (size_t)t * D * D);
            v4u g0;
            g0[0] = 1u;                                   // count = 1 valid D#
            g0[1] = 0u;                                   // lds_addr = 0 (wT)
            g0[2] = (unsigned int)(ga & 0xFFFFFFFFull);   // global_addr lo
            g0[3] = (unsigned int)((ga >> 32) & 0x1FFFFFFull) | (2u << 30); // hi | type=2
            v8i g1;
            g1[0] = (int)(1u << 16);        // data_size = 2 bytes (code 1)
            g1[1] = (int)(256u << 16);      // tensor_dim0 = 256 elems (bits 79:48 lo)
            g1[2] = (int)(256u << 16);      // tensor_dim1 = 256      (bits 111:80 lo)
            g1[3] = (int)(256u << 16);      // tile_dim0  = 256       (bits 127:112)
            g1[4] = 256;                    // tile_dim1  = 256       (bits 143:128)
            g1[5] = 256;                    // tensor_dim0_stride = 256 (bits 207:160 lo)
            g1[6] = 0;
            g1[7] = 0;
            v4i gz;  gz[0] = 0; gz[1] = 0; gz[2] = 0; gz[3] = 0;
            v8i gz8; gz8[0] = 0; gz8[1] = 0; gz8[2] = 0; gz8[3] = 0;
                     gz8[4] = 0; gz8[5] = 0; gz8[6] = 0; gz8[7] = 0;
            __builtin_amdgcn_tensor_load_to_lds(g0, g1, gz, gz, gz8, 0);
        }
        // ---- overlap: build per-type compacted row list while DMA flies ----
        if (tid == 0) *cntp = 0;
        __syncthreads();
        {
            int r = base + tid;
            if (r < rows && types[r] == t) {
                int p = atomicAdd(cntp, 1);
                list[p] = r;
            }
        }
        if (tid < 32) __builtin_amdgcn_s_wait_tensorcnt(0);
        __syncthreads();
        const int cnt = *cntp;
        const int ntiles = (cnt + 15) >> 4;
        for (int mt = 0; mt < ntiles; ++mt) {
            // stage 16-row A tile (fp32 -> bf16) into LDS, vectorized:
            // float4 global loads -> 4x cvt -> 8B DS store (4 iters/thread)
            for (int e = tid; e < (16 * D) / 4; e += 256) {
                int m = e >> 6;              // D/4 == 64 chunks per row
                int c = e & 63;
                int mi = mt * 16 + m;
                int row = (mi < cnt) ? list[mi] : list[0];
                const float4 xv = *(const float4*)(X + (size_t)row * D + c * 4);
                v4bf o;
                o[0] = (bf16_t)xv.x; o[1] = (bf16_t)xv.y;
                o[2] = (bf16_t)xv.z; o[3] = (bf16_t)xv.w;
                *(v4bf*)(aT + m * D + c * 4) = o;
            }
            __syncthreads();
            const int n0  = wave * 32;            // each wave: 32 output cols
            const int kh  = (lane >= 16) ? 16 : 0;
            const int l15 = lane & 15;
            v8f acc0 = {0.f, 0.f, 0.f, 0.f, 0.f, 0.f, 0.f, 0.f};
            v8f acc1 = acc0;
            for (int kc = 0; kc < D; kc += 32) {
                const int kb = kc + kh;
                v16bf a, b0, b1;
                const bf16_t* ap  = &aT[l15 * D + kb];
                const bf16_t* bp0 = &wT[(n0 + l15) * D + kb];
                const bf16_t* bp1 = &wT[(n0 + 16 + l15) * D + kb];
                #pragma unroll
                for (int i = 0; i < 16; ++i) {
                    a[i]  = ap[i];
                    b0[i] = bp0[i];
                    b1[i] = bp1[i];
                }
                acc0 = __builtin_amdgcn_wmma_f32_16x16x32_bf16(
                           false, a, false, b0, (short)0, acc0, false, false);
                acc1 = __builtin_amdgcn_wmma_f32_16x16x32_bf16(
                           false, a, false, b1, (short)0, acc1, false, false);
            }
            #pragma unroll
            for (int j = 0; j < 8; ++j) {
                int m  = j + ((lane >= 16) ? 8 : 0);
                int mi = mt * 16 + m;
                if (mi < cnt) {
                    size_t row = (size_t)list[mi];
                    Y[row * D + n0 + l15]      = (bf16_t)acc0[j];
                    Y[row * D + n0 + 16 + l15] = (bf16_t)acc1[j];
                }
            }
            __syncthreads();
        }
        __syncthreads();
    }
}

// ---------------------------------------------------------------------------
// Kernel 2: fused attention + inverse-propensity blend + per-type output
// projection (WMMA, bf16 pre-transposed Wa) + residual + LayerNorm.
// 16 targets per block, 8 waves.
// ---------------------------------------------------------------------------
__global__ void __launch_bounds__(256)
attn_out(const bf16_t* __restrict__ qb, const bf16_t* __restrict__ kbuf,
         const bf16_t* __restrict__ vbuf, const float* __restrict__ res,
         const int* __restrict__ ttypes, const float* __restrict__ psv,
         const unsigned char* __restrict__ mask, const bf16_t* __restrict__ waT,
         const float* __restrict__ gamma, const float* __restrict__ betaln,
         const float* __restrict__ skip, float* __restrict__ out, int B)
{
    extern __shared__ char smem[];
    float* qbuf  = (float*)smem;                 // [16*D]
    float* msg   = qbuf + 16 * D;                // [16*D]
    float* attnb = msg + 16 * D;                 // [8 waves * NH * NGB]
    float* mixed = attnb + 8 * NH * NGB;         // [16*D]
    int* list    = (int*)(mixed + 16 * D);       // [16]
    int* cntp    = list + 16;

    const int tid  = threadIdx.x;
    const int lane = tid & 31;
    const int wave = tid >> 5;
    const int b0   = blockIdx.x * 16;
    const float scale = 0.17677669529663687f;    // 1/sqrt(d_k=32)

    // ---- phase 1: each wave computes messages for 2 targets ----
    for (int it = 0; it < 2; ++it) {
        const int btl = wave * 2 + it;
        const int b   = b0 + btl;
        if (b >= B) continue;   // wave-uniform

        {   // stage q to LDS as float (one 16B bf16 vector per lane)
            v8bf qv = *(const v8bf*)(qb + (size_t)b * D + lane * 8);
            #pragma unroll
            for (int i = 0; i < 8; ++i)
                qbuf[btl * D + lane * 8 + i] = (float)qv[i];
        }

        // scores: lane owns neighbors (lane) and (lane+32), 8 heads each
        float sc[2][NH];
        #pragma unroll
        for (int sl = 0; sl < 2; ++sl)
            #pragma unroll
            for (int h = 0; h < NH; ++h) sc[sl][h] = 0.f;

        for (int sl = 0; sl < 2; ++sl) {
            const int n = lane + sl * 32;
            const bf16_t* krow = kbuf + ((size_t)b * NGB + n) * D;
            for (int c = 0; c < 32; ++c) {
                const int h  = c >> 2;
                const int d0 = c * 8;
                v8bf kv = *(const v8bf*)(krow + d0);
                float s = 0.f;
                #pragma unroll
                for (int i = 0; i < 8; ++i)
                    s += qbuf[btl * D + d0 + i] * (float)kv[i];
                sc[sl][h] += s;
            }
            bool mk = mask[(size_t)b * NGB + n] != 0;
            #pragma unroll
            for (int h = 0; h < NH; ++h)
                sc[sl][h] = mk ? -1e10f : sc[sl][h] * scale;
        }

        // inverse-propensity distribution (over all 64 neighbors)
        float ip0 = 1.f / psv[(size_t)b * NGB + lane];
        float ip1 = 1.f / psv[(size_t)b * NGB + lane + 32];
        float tot = ip0 + ip1;
        #pragma unroll
        for (int off = 16; off >= 1; off >>= 1) tot += __shfl_xor(tot, off, 32);

        // per-head masked softmax + blend -> LDS
        #pragma unroll
        for (int h = 0; h < NH; ++h) {
            float mx = fmaxf(sc[0][h], sc[1][h]);
            #pragma unroll
            for (int off = 16; off >= 1; off >>= 1)
                mx = fmaxf(mx, __shfl_xor(mx, off, 32));
            float e0 = __expf(sc[0][h] - mx);
            float e1 = __expf(sc[1][h] - mx);
            float sm = e0 + e1;
            #pragma unroll
            for (int off = 16; off >= 1; off >>= 1) sm += __shfl_xor(sm, off, 32);
            float inv = 1.f / sm;
            attnb[wave * (NH * NGB) + h * NGB + lane] =
                e0 * inv * (1.f - PS_ALPHA) + (ip0 / tot) * PS_ALPHA;
            attnb[wave * (NH * NGB) + h * NGB + lane + 32] =
                e1 * inv * (1.f - PS_ALPHA) + (ip1 / tot) * PS_ALPHA;
        }

        // message: lane owns dims [lane*8, lane*8+8), all within head lane>>2
        const int h  = lane >> 2;
        const int d0 = lane * 8;
        float acc[8] = {0.f, 0.f, 0.f, 0.f, 0.f, 0.f, 0.f, 0.f};
        for (int n = 0; n < NGB; ++n) {
            float a = attnb[wave * (NH * NGB) + h * NGB + n];
            v8bf vv = *(const v8bf*)(vbuf + ((size_t)b * NGB + n) * D + d0);
            #pragma unroll
            for (int i = 0; i < 8; ++i) acc[i] += a * (float)vv[i];
        }
        #pragma unroll
        for (int i = 0; i < 8; ++i) msg[btl * D + d0 + i] = acc[i];
    }
    __syncthreads();

    // ---- phase 2: per-type WMMA output projection + residual + LN ----
    for (int t = 0; t < TT; ++t) {
        if (tid == 0) *cntp = 0;
        __syncthreads();
        if (tid < 16 && (b0 + tid) < B && ttypes[b0 + tid] == t) {
            int p = atomicAdd(cntp, 1);
            list[p] = tid;
        }
        __syncthreads();
        const int cnt = *cntp;
        if (cnt > 0) {   // uniform branch: EXEC all-ones at WMMA
            const int n0  = wave * 32;
            const int kh  = (lane >= 16) ? 16 : 0;
            const int l15 = lane & 15;
            const int btl = (l15 < cnt) ? list[l15] : list[0];
            v8f acc0 = {0.f, 0.f, 0.f, 0.f, 0.f, 0.f, 0.f, 0.f};
            v8f acc1 = acc0;
            const bf16_t* Wt = waT + (size_t)t * D * D;
            for (int kc = 0; kc < D; kc += 32) {
                const int kbk = kc + kh;
                v16bf a, w0, w1;
                const bf16_t* wp0 = &Wt[(size_t)(n0 + l15) * D + kbk];
                const bf16_t* wp1 = &Wt[(size_t)(n0 + 16 + l15) * D + kbk];
                #pragma unroll
                for (int i = 0; i < 16; ++i) {
                    a[i]  = (bf16_t)msg[btl * D + kbk + i];
                    w0[i] = wp0[i];
                    w1[i] = wp1[i];
                }
                acc0 = __builtin_amdgcn_wmma_f32_16x16x32_bf16(
                           false, a, false, w0, (short)0, acc0, false, false);
                acc1 = __builtin_amdgcn_wmma_f32_16x16x32_bf16(
                           false, a, false, w1, (short)0, acc1, false, false);
            }
            const float bt = 1.f / (1.f + __expf(-skip[t]));
            #pragma unroll
            for (int j = 0; j < 8; ++j) {
                int m = j + ((lane >= 16) ? 8 : 0);
                if (m < cnt) {
                    int bl   = list[m];
                    size_t b = (size_t)(b0 + bl);
                    int c0   = n0 + l15;
                    mixed[bl * D + c0] =
                        acc0[j] * bt + res[b * D + c0] * (1.f - bt);
                    mixed[bl * D + c0 + 16] =
                        acc1[j] * bt + res[b * D + c0 + 16] * (1.f - bt);
                }
            }
        }
        __syncthreads();
        // LayerNorm: each wave normalizes up to 2 rows of this type
        for (int mi = wave * 2; mi < wave * 2 + 2; ++mi) {
            if (mi < cnt) {
                int bl = list[mi];
                const int d0 = lane * 8;
                float s = 0.f, s2 = 0.f;
                #pragma unroll
                for (int i = 0; i < 8; ++i) {
                    float x = mixed[bl * D + d0 + i];
                    s += x; s2 += x * x;
                }
                #pragma unroll
                for (int off = 16; off >= 1; off >>= 1) {
                    s  += __shfl_xor(s, off, 32);
                    s2 += __shfl_xor(s2, off, 32);
                }
                float mu  = s * (1.f / D);
                float var = s2 * (1.f / D) - mu * mu;
                float rs  = rsqrtf(var + LN_EPS);
                size_t b  = (size_t)(b0 + bl);
                #pragma unroll
                for (int i = 0; i < 8; ++i) {
                    int d = d0 + i;
                    out[b * D + d] = (mixed[bl * D + d] - mu) * rs *
                                     gamma[t * D + d] + betaln[t * D + d];
                }
            }
        }
        __syncthreads();
    }
}

// ---------------------------------------------------------------------------
extern "C" void kernel_launch(void* const* d_in, const int* in_sizes, int n_in,
                              void* d_out, int out_size, void* d_ws, size_t ws_size,
                              hipStream_t stream)
{
    const float* tgt  = (const float*)d_in[0];
    const int*   ttyp = (const int*)d_in[1];
    const float* src  = (const float*)d_in[2];
    const int*   styp = (const int*)d_in[3];
    const float* psv  = (const float*)d_in[4];
    const unsigned char* mask = (const unsigned char*)d_in[5];
    const float* Wq   = (const float*)d_in[6];
    const float* Wk   = (const float*)d_in[7];
    const float* Wv   = (const float*)d_in[8];
    const float* Wa   = (const float*)d_in[9];
    const float* gmm  = (const float*)d_in[10];
    const float* btl  = (const float*)d_in[11];
    const float* skp  = (const float*)d_in[12];
    float* out = (float*)d_out;

    const int B  = in_sizes[1];      // tgt_node_types count
    const int SR = B * NGB;          // total source rows

    const size_t qbytes  = (size_t)B * D * sizeof(bf16_t);
    const size_t kvbytes = (size_t)SR * D * sizeof(bf16_t);
    const size_t wbytes  = (size_t)TT * D * D * sizeof(bf16_t);
    if (ws_size < qbytes + 2 * kvbytes + 4 * wbytes) return;
    char* p = (char*)d_ws;
    bf16_t* qbuf = (bf16_t*)p;             p += qbytes;
    bf16_t* kbuf = (bf16_t*)p;             p += kvbytes;
    bf16_t* vbuf = (bf16_t*)p;             p += kvbytes;
    bf16_t* wqT  = (bf16_t*)p;             p += wbytes;
    bf16_t* wkT  = (bf16_t*)p;             p += wbytes;
    bf16_t* wvT  = (bf16_t*)p;             p += wbytes;
    bf16_t* waT  = (bf16_t*)p;             p += wbytes;

    const int smem1 = D * D * 2 + 16 * D * 2 + (256 + 8) * 4;           // ~140 KB
    const int smem2 = (16 * D * 3 + 8 * NH * NGB) * 4 + (16 + 8) * 4;   // ~64 KB
    (void)hipFuncSetAttribute(reinterpret_cast<const void*>(project_rows),
                              hipFuncAttributeMaxDynamicSharedMemorySize, smem1);
    (void)hipFuncSetAttribute(reinterpret_cast<const void*>(attn_out),
                              hipFuncAttributeMaxDynamicSharedMemorySize, smem2);

    const int wtotal = TT * D * D;
    const int wgrid  = (wtotal + 255) / 256;
    transpose_cvt_weights<<<wgrid, 256, 0, stream>>>(Wq, wqT, wtotal);
    transpose_cvt_weights<<<wgrid, 256, 0, stream>>>(Wk, wkT, wtotal);
    transpose_cvt_weights<<<wgrid, 256, 0, stream>>>(Wv, wvT, wtotal);
    transpose_cvt_weights<<<wgrid, 256, 0, stream>>>(Wa, waT, wtotal);

    project_rows<<<(B + 255) / 256, 256, smem1, stream>>>(tgt, ttyp, wqT, qbuf, B);
    project_rows<<<(SR + 255) / 256, 256, smem1, stream>>>(src, styp, wkT, kbuf, SR);
    project_rows<<<(SR + 255) / 256, 256, smem1, stream>>>(src, styp, wvT, vbuf, SR);
    attn_out<<<(B + 15) / 16, 256, smem2, stream>>>(qbuf, kbuf, vbuf, tgt, ttyp, psv,
                                                    mask, waT, gmm, btl, skp, out, B);
}
